// Net_72662256713812
// MI455X (gfx1250) — compile-verified, matrix operand order checked by reading, not run on hardware
//
#include <hip/hip_runtime.h>
#include <math.h>

// ---------------------------------------------------------------------------
// Problem constants (from reference)
// ---------------------------------------------------------------------------
#define KB 4
#define KT 512
#define KIN 80
#define KINP 96              // K-padded to multiple of 32
#define KOUT 4096
#define KD 512
#define KHID 1024
#define KE 4
#define KH 8
#define KM 64
#define KDH (KD / KH)        // 64
#define KLB 4
#define KSL 2
#define KSR 1
#define KBT (KB * KT)        // 2048 tokens
#define KTM (KT + KM)        // 576 keys
#define KBTOUT ((size_t)KBT * KOUT)

typedef __attribute__((ext_vector_type(16))) __bf16 v16bf;
typedef __attribute__((ext_vector_type(8)))  float  v8f;

// ---------------------------------------------------------------------------
// WMMA helpers (per CDNA5 ISA 7.12.2 wave32 layouts)
// A 16x32 bf16: lanes 0-15 hold row M=lane, K in {0..7,16..23}; lanes 16-31
// hold K in {8..15,24..31}. B mirrored with N=lane%16. C/D: vgpr r ->
// M = r + 8*(lane/16), N = lane%16.
// ---------------------------------------------------------------------------
__device__ __forceinline__ v8f wmma_bf16(v16bf a, v16bf b, v8f c) {
  return __builtin_amdgcn_wmma_f32_16x16x32_bf16(false, a, false, b, (short)0, c,
                                                 false, false);
}

__device__ __forceinline__ unsigned pack_bf16x2(float lo, float hi) {
  union { __bf16 h[2]; unsigned u; } x;
  x.h[0] = (__bf16)lo;
  x.h[1] = (__bf16)hi;
  return x.u;
}

// A fragment: branch-free, two 8-float contiguous chunks via float4 loads.
__device__ __forceinline__ v16bf load_a_frag(const float* __restrict__ A, int lda,
                                             int row, int kb, int khalf) {
  const float* r = A + (size_t)row * lda + kb + khalf * 8;
  float4 a0 = *(const float4*)(r + 0);
  float4 a1 = *(const float4*)(r + 4);
  float4 a2 = *(const float4*)(r + 16);
  float4 a3 = *(const float4*)(r + 20);
  v16bf f;
  f[0] = (__bf16)a0.x;  f[1] = (__bf16)a0.y;  f[2] = (__bf16)a0.z;  f[3] = (__bf16)a0.w;
  f[4] = (__bf16)a1.x;  f[5] = (__bf16)a1.y;  f[6] = (__bf16)a1.z;  f[7] = (__bf16)a1.w;
  f[8] = (__bf16)a2.x;  f[9] = (__bf16)a2.y;  f[10] = (__bf16)a2.z; f[11] = (__bf16)a2.w;
  f[12] = (__bf16)a3.x; f[13] = (__bf16)a3.y; f[14] = (__bf16)a3.z; f[15] = (__bf16)a3.w;
  return f;
}

// B fragment from LDS (packed bf16x2, stride 20 dwords per column, 16B aligned).
#define LDSB_STRIDE 20
__device__ __forceinline__ v16bf load_b_frag_lds(const unsigned* __restrict__ ldsB,
                                                 int n, int khalf) {
  union { v16bf v; uint4 q[2]; } u;
  const unsigned* p = ldsB + n * LDSB_STRIDE + khalf * 4;
  u.q[0] = *(const uint4*)(p + 0);
  u.q[1] = *(const uint4*)(p + 8);
  return u.v;
}

// Cooperative stage of B[kb..kb+31][n0..n0+63] (f32, row-major) into LDS bf16x2.
__device__ __forceinline__ void stage_b(const float* __restrict__ Bm, int N, int kb,
                                        int n0, unsigned* __restrict__ ldsB, int tid) {
#pragma unroll
  for (int u = 0; u < 4; ++u) {
    int id = tid + u * 256;
    int n = id & 63;
    int k2 = id >> 6;
    size_t g = (size_t)(kb + 2 * k2) * N + (n0 + n);
    ldsB[n * LDSB_STRIDE + k2] = pack_bf16x2(Bm[g], Bm[g + N]);
  }
}

// ---------------------------------------------------------------------------
// Generic GEMM: C[M,N] = act(A[M,K] @ B[K,N] (+bias)).
// 256 thr = 8 wave32. Block tile 256(M) x 64(N); wave tile 32x64, 8 WMMA/step.
// K must be a multiple of 32; M a multiple of 256 here.
// ---------------------------------------------------------------------------
template <bool RELU, bool BIAS>
__global__ __launch_bounds__(256) void gemm_bf16_kernel(
    const float* __restrict__ A, const float* __restrict__ Bm,
    const float* __restrict__ bias, float* __restrict__ C, int M, int N, int K) {
  __shared__ unsigned ldsB[64 * LDSB_STRIDE];
  const int tid = threadIdx.x;
  const int lane = tid & 31;
  const int wave = tid >> 5;
  const int lrow = lane & 15;
  const int khalf = lane >> 4;
  const int mbase = blockIdx.y * 256 + wave * 32;
  const int n0 = blockIdx.x * 64;

  v8f acc[2][4] = {};
  for (int kb = 0; kb < K; kb += 32) {
    stage_b(Bm, N, kb, n0, ldsB, tid);
    __syncthreads();
    v16bf a0 = load_a_frag(A, K, mbase + lrow, kb, khalf);
    v16bf a1 = load_a_frag(A, K, mbase + 16 + lrow, kb, khalf);
    v16bf b0 = load_b_frag_lds(ldsB, 0 * 16 + lrow, khalf);
    v16bf b1 = load_b_frag_lds(ldsB, 1 * 16 + lrow, khalf);
    v16bf b2 = load_b_frag_lds(ldsB, 2 * 16 + lrow, khalf);
    v16bf b3 = load_b_frag_lds(ldsB, 3 * 16 + lrow, khalf);
    acc[0][0] = wmma_bf16(a0, b0, acc[0][0]);
    acc[0][1] = wmma_bf16(a0, b1, acc[0][1]);
    acc[0][2] = wmma_bf16(a0, b2, acc[0][2]);
    acc[0][3] = wmma_bf16(a0, b3, acc[0][3]);
    acc[1][0] = wmma_bf16(a1, b0, acc[1][0]);
    acc[1][1] = wmma_bf16(a1, b1, acc[1][1]);
    acc[1][2] = wmma_bf16(a1, b2, acc[1][2]);
    acc[1][3] = wmma_bf16(a1, b3, acc[1][3]);
    __syncthreads();
  }
#pragma unroll
  for (int h = 0; h < 2; ++h) {
#pragma unroll
    for (int t = 0; t < 4; ++t) {
      const int col = n0 + t * 16 + lrow;
      const float bv = BIAS ? bias[col] : 0.0f;
#pragma unroll
      for (int r = 0; r < 8; ++r) {
        const int row = mbase + h * 16 + khalf * 8 + r;
        float v = acc[h][t][r] + bv;
        if (RELU) v = fmaxf(v, 0.0f);
        C[(size_t)row * N + col] = v;
      }
    }
  }
}

// ---------------------------------------------------------------------------
// MoE expert GEMM 1: gathered rows of X -> h = relu(X[idx] @ W1[e] + b1[e]).
// Compacted output rows at [offsets[e] + local].  grid.z = expert.
// ---------------------------------------------------------------------------
__global__ __launch_bounds__(256) void moe_gemm1_kernel(
    const float* __restrict__ X, const float* __restrict__ W1,
    const float* __restrict__ B1, const int* __restrict__ idxmap,
    const int* __restrict__ offsets, float* __restrict__ Hc, int Kin) {
  const int e = blockIdx.z;
  const int start = offsets[e];
  const int cnt = offsets[e + 1] - start;
  if ((int)blockIdx.y * 256 >= cnt) return;      // block-uniform early out
  __shared__ unsigned ldsB[64 * LDSB_STRIDE];
  const int tid = threadIdx.x;
  const int lane = tid & 31;
  const int wave = tid >> 5;
  const int lrow = lane & 15;
  const int khalf = lane >> 4;
  const int rbase = blockIdx.y * 256 + wave * 32;
  const int n0 = blockIdx.x * 64;
  const float* W = W1 + (size_t)e * Kin * KHID;
  const float* bias = B1 + (size_t)e * KHID;

  int glo = rbase + lrow;
  int ghi = rbase + 16 + lrow;
  glo = (glo < cnt) ? glo : (cnt - 1);
  ghi = (ghi < cnt) ? ghi : (cnt - 1);
  const int rlo = idxmap[start + glo];
  const int rhi = idxmap[start + ghi];

  v8f acc[2][4] = {};
  for (int kb = 0; kb < Kin; kb += 32) {
    stage_b(W, KHID, kb, n0, ldsB, tid);
    __syncthreads();
    v16bf a0 = load_a_frag(X, Kin, rlo, kb, khalf);
    v16bf a1 = load_a_frag(X, Kin, rhi, kb, khalf);
    v16bf b0 = load_b_frag_lds(ldsB, 0 * 16 + lrow, khalf);
    v16bf b1 = load_b_frag_lds(ldsB, 1 * 16 + lrow, khalf);
    v16bf b2 = load_b_frag_lds(ldsB, 2 * 16 + lrow, khalf);
    v16bf b3 = load_b_frag_lds(ldsB, 3 * 16 + lrow, khalf);
    acc[0][0] = wmma_bf16(a0, b0, acc[0][0]);
    acc[0][1] = wmma_bf16(a0, b1, acc[0][1]);
    acc[0][2] = wmma_bf16(a0, b2, acc[0][2]);
    acc[0][3] = wmma_bf16(a0, b3, acc[0][3]);
    acc[1][0] = wmma_bf16(a1, b0, acc[1][0]);
    acc[1][1] = wmma_bf16(a1, b1, acc[1][1]);
    acc[1][2] = wmma_bf16(a1, b2, acc[1][2]);
    acc[1][3] = wmma_bf16(a1, b3, acc[1][3]);
    __syncthreads();
  }
#pragma unroll
  for (int h = 0; h < 2; ++h) {
#pragma unroll
    for (int t = 0; t < 4; ++t) {
      const int col = n0 + t * 16 + lrow;
      const float bv = bias[col];
#pragma unroll
      for (int r = 0; r < 8; ++r) {
        const int m = rbase + h * 16 + khalf * 8 + r;
        if (m < cnt) {
          Hc[(size_t)(start + m) * KHID + col] = fmaxf(acc[h][t][r] + bv, 0.0f);
        }
      }
    }
  }
}

// ---------------------------------------------------------------------------
// MoE expert GEMM 2: p[token] = gate[token] * (h_compact @ W2[e]), scattered.
// ---------------------------------------------------------------------------
__global__ __launch_bounds__(256) void moe_gemm2_kernel(
    const float* __restrict__ Hc, const float* __restrict__ W2,
    const int* __restrict__ idxmap, const int* __restrict__ offsets,
    const float* __restrict__ gate, float* __restrict__ P) {
  const int e = blockIdx.z;
  const int start = offsets[e];
  const int cnt = offsets[e + 1] - start;
  if ((int)blockIdx.y * 256 >= cnt) return;      // block-uniform early out
  __shared__ unsigned ldsB[64 * LDSB_STRIDE];
  const int tid = threadIdx.x;
  const int lane = tid & 31;
  const int wave = tid >> 5;
  const int lrow = lane & 15;
  const int khalf = lane >> 4;
  const int rbase = blockIdx.y * 256 + wave * 32;
  const int n0 = blockIdx.x * 64;
  const float* W = W2 + (size_t)e * KHID * KD;

  int glo = rbase + lrow;
  int ghi = rbase + 16 + lrow;
  glo = (glo < cnt) ? glo : (cnt - 1);
  ghi = (ghi < cnt) ? ghi : (cnt - 1);
  const int rlo = start + glo;
  const int rhi = start + ghi;

  v8f acc[2][4] = {};
  for (int kb = 0; kb < KHID; kb += 32) {
    stage_b(W, KD, kb, n0, ldsB, tid);
    __syncthreads();
    v16bf a0 = load_a_frag(Hc, KHID, rlo, kb, khalf);
    v16bf a1 = load_a_frag(Hc, KHID, rhi, kb, khalf);
    v16bf b0 = load_b_frag_lds(ldsB, 0 * 16 + lrow, khalf);
    v16bf b1 = load_b_frag_lds(ldsB, 1 * 16 + lrow, khalf);
    v16bf b2 = load_b_frag_lds(ldsB, 2 * 16 + lrow, khalf);
    v16bf b3 = load_b_frag_lds(ldsB, 3 * 16 + lrow, khalf);
    acc[0][0] = wmma_bf16(a0, b0, acc[0][0]);
    acc[0][1] = wmma_bf16(a0, b1, acc[0][1]);
    acc[0][2] = wmma_bf16(a0, b2, acc[0][2]);
    acc[0][3] = wmma_bf16(a0, b3, acc[0][3]);
    acc[1][0] = wmma_bf16(a1, b0, acc[1][0]);
    acc[1][1] = wmma_bf16(a1, b1, acc[1][1]);
    acc[1][2] = wmma_bf16(a1, b2, acc[1][2]);
    acc[1][3] = wmma_bf16(a1, b3, acc[1][3]);
    __syncthreads();
  }
#pragma unroll
  for (int h = 0; h < 2; ++h) {
#pragma unroll
    for (int r = 0; r < 8; ++r) {
      const int m = rbase + h * 16 + khalf * 8 + r;
      if (m < cnt) {
        const int tok = idxmap[start + m];
        const float g = gate[tok];
#pragma unroll
        for (int t = 0; t < 4; ++t) {
          const int col = n0 + t * 16 + lrow;
          P[(size_t)tok * KD + col] = g * acc[h][t][r];
        }
      }
    }
  }
}

// ---------------------------------------------------------------------------
// Router: per token (one wave): gates = softmax(embed @ Rw), top-1, masked
// gates stored for deterministic importance reduction.
// ---------------------------------------------------------------------------
__global__ void router_kernel(const float* __restrict__ embed,
                              const float* __restrict__ Rw,
                              int* __restrict__ expert_id,
                              float* __restrict__ gatev,
                              float* __restrict__ gates4,
                              int* __restrict__ counts,
                              const int* __restrict__ seq_len) {
  const int lane = threadIdx.x & 31;
  const int wave = threadIdx.x >> 5;
  const int t = blockIdx.x * 8 + wave;
  if (t >= KBT) return;
  const float* em = embed + (size_t)t * KD;
  float r0 = 0.f, r1 = 0.f, r2 = 0.f, r3 = 0.f;
  for (int d = lane; d < KD; d += 32) {
    float x = em[d];
    r0 += x * Rw[d * KE + 0];
    r1 += x * Rw[d * KE + 1];
    r2 += x * Rw[d * KE + 2];
    r3 += x * Rw[d * KE + 3];
  }
#pragma unroll
  for (int off = 16; off; off >>= 1) {
    r0 += __shfl_xor(r0, off, 32);
    r1 += __shfl_xor(r1, off, 32);
    r2 += __shfl_xor(r2, off, 32);
    r3 += __shfl_xor(r3, off, 32);
  }
  if (lane == 0) {
    float mx = fmaxf(fmaxf(r0, r1), fmaxf(r2, r3));
    float e0 = __expf(r0 - mx), e1 = __expf(r1 - mx);
    float e2 = __expf(r2 - mx), e3 = __expf(r3 - mx);
    float s = e0 + e1 + e2 + e3;
    float g0 = e0 / s, g1 = e1 / s, g2 = e2 / s, g3 = e3 / s;
    int amax = 0;
    float gm = g0;
    if (g1 > gm) { gm = g1; amax = 1; }
    if (g2 > gm) { gm = g2; amax = 2; }
    if (g3 > gm) { gm = g3; amax = 3; }
    expert_id[t] = amax;
    gatev[t] = gm;
    atomicAdd(&counts[amax], 1);
    int b = t / KT, tt = t % KT;
    float valid = (tt < seq_len[b]) ? 1.0f : 0.0f;
    gates4[t * 4 + 0] = g0 * valid;
    gates4[t * 4 + 1] = g1 * valid;
    gates4[t * 4 + 2] = g2 * valid;
    gates4[t * 4 + 3] = g3 * valid;
  }
}

__global__ void scan_kernel(const int* __restrict__ counts, int* __restrict__ offsets) {
  if (threadIdx.x == 0) {
    offsets[0] = 0;
    for (int e = 0; e < KE; ++e) offsets[e + 1] = offsets[e] + counts[e];
  }
}

__global__ void scatter_kernel(const int* __restrict__ expert_id,
                               const int* __restrict__ offsets,
                               int* __restrict__ cursor, int* __restrict__ idxmap) {
  int t = blockIdx.x * 256 + threadIdx.x;
  if (t >= KBT) return;
  int e = expert_id[t];
  int pos = atomicAdd(&cursor[e], 1);
  idxmap[offsets[e] + pos] = t;
}

// Deterministic per-layer importance-cv^2 aux.
__global__ void aux_kernel(const float* __restrict__ gates4,
                           float* __restrict__ auxbuf, int layer) {
  __shared__ float sh[256 * 4];
  const int tid = threadIdx.x;
  float p0 = 0.f, p1 = 0.f, p2 = 0.f, p3 = 0.f;
  for (int t = tid; t < KBT; t += 256) {
    p0 += gates4[t * 4 + 0];
    p1 += gates4[t * 4 + 1];
    p2 += gates4[t * 4 + 2];
    p3 += gates4[t * 4 + 3];
  }
  sh[tid * 4 + 0] = p0; sh[tid * 4 + 1] = p1;
  sh[tid * 4 + 2] = p2; sh[tid * 4 + 3] = p3;
  __syncthreads();
  for (int s = 128; s; s >>= 1) {
    if (tid < s) {
#pragma unroll
      for (int e = 0; e < 4; ++e) sh[tid * 4 + e] += sh[(tid + s) * 4 + e];
    }
    __syncthreads();
  }
  if (tid == 0) {
    float i0 = sh[0], i1 = sh[1], i2 = sh[2], i3 = sh[3];
    float mean = (i0 + i1 + i2 + i3) * 0.25f;
    float d0 = i0 - mean, d1 = i1 - mean, d2 = i2 - mean, d3 = i3 - mean;
    float var = (d0 * d0 + d1 * d1 + d2 * d2 + d3 * d3) * 0.25f;
    auxbuf[layer] = var / (mean * mean + 1e-9f);
  }
}

__global__ void sum_aux_kernel(const float* __restrict__ auxbuf, float* __restrict__ out) {
  if (threadIdx.x == 0) {
    float s = 0.f;
    for (int i = 0; i < 8; ++i) s += auxbuf[i];
    out[0] = s;
  }
}

// ---------------------------------------------------------------------------
// FSMN depthwise memory: m[t] = p[t] + sum_k fb[k]*p[t-2(k+1)] + fa*p[t+1] (+xi)
// ---------------------------------------------------------------------------
__global__ void fsmn_kernel(const float* __restrict__ P, const float* __restrict__ fb,
                            const float* __restrict__ fa, const float* __restrict__ Xi,
                            float* __restrict__ Out, int skip) {
  int i = blockIdx.x * 256 + threadIdx.x;
  if (i >= KBT * KD) return;
  int d = i % KD;
  int t = (i / KD) % KT;
  float m = P[i];
#pragma unroll
  for (int k = 0; k < KLB; ++k) {
    int off = (k + 1) * KSL;
    if (t >= off) m += fb[k * KD + d] * P[i - off * KD];
  }
  if (t + KSR < KT) m += fa[d] * P[i + KSR * KD];
  if (skip) m += Xi[i];
  Out[i] = m;
}

// Sinusoidal positional encoding, computed in place.
__global__ void pe_kernel(float* __restrict__ cur) {
  int i = blockIdx.x * 256 + threadIdx.x;
  if (i >= KBT * KD) return;
  int d = i % KD;
  int t = (i / KD) % KT;
  float freq = __expf(-(float)(d & ~1) * (9.210340371976184f / (float)KD));
  float ang = (float)t * freq;
  cur[i] += (d & 1) ? __cosf(ang) : __sinf(ang);
}

// Concatenate projected K/V with persistent memory slots.
__global__ void build_kv_kernel(const float* __restrict__ kp, const float* __restrict__ vp,
                                const float* __restrict__ mk, const float* __restrict__ mv,
                                float* __restrict__ kfull, float* __restrict__ vfull) {
  int i = blockIdx.x * 256 + threadIdx.x;
  if (i >= KB * KTM * KD) return;
  int d = i % KD;
  int s = (i / KD) % KTM;
  int b = i / (KTM * KD);
  if (s < KT) {
    size_t src = (size_t)(b * KT + s) * KD + d;
    kfull[i] = kp[src];
    vfull[i] = vp[src];
  } else {
    size_t src = (size_t)(s - KT) * KD + d;
    kfull[i] = mk[src];
    vfull[i] = mv[src];
  }
}

// Streaming attention: one wave per (b,h,t); online softmax over 576 keys.
__global__ void attention_kernel(const float* __restrict__ q,
                                 const float* __restrict__ kfull,
                                 const float* __restrict__ vfull,
                                 const int* __restrict__ seq_len,
                                 float* __restrict__ obuf) {
  const int lane = threadIdx.x & 31;
  const int wave = threadIdx.x >> 5;
  int gw = blockIdx.x * 8 + wave;           // 0 .. B*H*T-1
  if (gw >= KB * KH * KT) return;
  const int b = gw / (KH * KT);
  const int rem = gw % (KH * KT);
  const int h = rem / KT;
  const int t = rem % KT;
  const int sl = seq_len[b];

  const float* qrow = q + (size_t)(b * KT + t) * KD + h * KDH;
  const float q0 = qrow[lane * 2 + 0];
  const float q1 = qrow[lane * 2 + 1];

  float mmax = -3.4e38f, ssum = 0.f, o0 = 0.f, o1 = 0.f;
  for (int s = 0; s < KTM; ++s) {
    const float* krow = kfull + ((size_t)b * KTM + s) * KD + h * KDH;
    float part = q0 * krow[lane * 2 + 0] + q1 * krow[lane * 2 + 1];
#pragma unroll
    for (int off = 16; off; off >>= 1) part += __shfl_xor(part, off, 32);
    bool ok = (s >= KT) || (s < sl);
    float score = ok ? part * 0.125f : -1e9f;
    float nm = fmaxf(mmax, score);
    float corr = __expf(mmax - nm);
    float w = __expf(score - nm);
    const float* vrow = vfull + ((size_t)b * KTM + s) * KD + h * KDH;
    ssum = ssum * corr + w;
    o0 = o0 * corr + w * vrow[lane * 2 + 0];
    o1 = o1 * corr + w * vrow[lane * 2 + 1];
    mmax = nm;
  }
  float inv = 1.0f / ssum;
  float* orow = obuf + (size_t)(b * KT + t) * KD + h * KDH;
  orow[lane * 2 + 0] = o0 * inv;
  orow[lane * 2 + 1] = o1 * inv;
}

// Residual add + layernorm: cur = LN(cur + oproj) * g + b ; wave per row.
__global__ void residual_ln_kernel(float* __restrict__ cur, const float* __restrict__ oproj,
                                   const float* __restrict__ g, const float* __restrict__ bta) {
  const int lane = threadIdx.x & 31;
  const int wave = threadIdx.x >> 5;
  int row = blockIdx.x * 8 + wave;
  if (row >= KBT) return;
  const float* a = cur + (size_t)row * KD;
  const float* o = oproj + (size_t)row * KD;
  float vals[16];
  float s = 0.f, ss = 0.f;
#pragma unroll
  for (int j = 0; j < 16; ++j) {
    int d = lane + j * 32;
    float v = a[d] + o[d];
    vals[j] = v;
    s += v;
    ss += v * v;
  }
#pragma unroll
  for (int off = 16; off; off >>= 1) {
    s += __shfl_xor(s, off, 32);
    ss += __shfl_xor(ss, off, 32);
  }
  float mu = s * (1.0f / KD);
  float var = ss * (1.0f / KD) - mu * mu;
  float inv = rsqrtf(var + 1e-5f);
#pragma unroll
  for (int j = 0; j < 16; ++j) {
    int d = lane + j * 32;
    cur[(size_t)row * KD + d] = (vals[j] - mu) * inv * g[d] + bta[d];
  }
}

__global__ void zero_ints_kernel(int* __restrict__ p, int n) {
  int i = blockIdx.x * 64 + threadIdx.x;
  if (i < n) p[i] = 0;
}

// Pad token matrix rows from KIN to KINP columns (zero fill).
__global__ void pad_rows_kernel(const float* __restrict__ src, float* __restrict__ dst) {
  int i = blockIdx.x * 256 + threadIdx.x;
  if (i >= KBT * KINP) return;
  int r = i / KINP, c = i % KINP;
  dst[i] = (c < KIN) ? src[(size_t)r * KIN + c] : 0.0f;
}

// Pad a [KIN,N] weight to [KINP,N] (zero tail rows); same linear prefix.
__global__ void pad_tail_kernel(const float* __restrict__ src, float* __restrict__ dst,
                                int ncopy, int ntotal) {
  int i = blockIdx.x * 256 + threadIdx.x;
  if (i >= ntotal) return;
  dst[i] = (i < ncopy) ? src[i] : 0.0f;
}

// Pad f0_w1 [E,KIN,HID] -> [E,KINP,HID].
__global__ void pad_f0w1_kernel(const float* __restrict__ src, float* __restrict__ dst) {
  int i = blockIdx.x * 256 + threadIdx.x;
  const int per = KINP * KHID;
  if (i >= KE * per) return;
  int e = i / per;
  int r = i % per;
  dst[i] = (r < KIN * KHID) ? src[(size_t)e * KIN * KHID + r] : 0.0f;
}

// ---------------------------------------------------------------------------
// Host-side orchestration
// ---------------------------------------------------------------------------
static void launch_gemm(const float* A, const float* Bm, const float* bias, float* C,
                        int M, int N, int K, bool relu, hipStream_t s) {
  dim3 grid(N / 64, M / 256);
  dim3 blk(256);
  if (bias) {
    if (relu) gemm_bf16_kernel<true, true><<<grid, blk, 0, s>>>(A, Bm, bias, C, M, N, K);
    else      gemm_bf16_kernel<false, true><<<grid, blk, 0, s>>>(A, Bm, bias, C, M, N, K);
  } else {
    if (relu) gemm_bf16_kernel<true, false><<<grid, blk, 0, s>>>(A, Bm, bias, C, M, N, K);
    else      gemm_bf16_kernel<false, false><<<grid, blk, 0, s>>>(A, Bm, bias, C, M, N, K);
  }
}

extern "C" void kernel_launch(void* const* d_in, const int* in_sizes, int n_in,
                              void* d_out, int out_size, void* d_ws, size_t ws_size,
                              hipStream_t stream) {
  (void)in_sizes; (void)n_in; (void)out_size; (void)ws_size;

  const float* x        = (const float*)d_in[0];
  const int*   seq_len  = (const int*)  d_in[1];
  const float* emb_w1   = (const float*)d_in[2];
  const float* emb_b1   = (const float*)d_in[3];
  const float* emb_w2   = (const float*)d_in[4];
  const float* emb_wo   = (const float*)d_in[5];
  const float* f0_w1    = (const float*)d_in[6];
  const float* f0_b1    = (const float*)d_in[7];
  const float* f0_w2    = (const float*)d_in[8];
  const float* f0_rout  = (const float*)d_in[9];
  const float* f0_fb    = (const float*)d_in[10];
  const float* f0_fa    = (const float*)d_in[11];
  const float* f_w1     = (const float*)d_in[12];
  const float* f_b1     = (const float*)d_in[13];
  const float* f_w2     = (const float*)d_in[14];
  const float* f_rout   = (const float*)d_in[15];
  const float* f_fb     = (const float*)d_in[16];
  const float* f_fa     = (const float*)d_in[17];
  const float* attn_wq  = (const float*)d_in[18];
  const float* attn_wk  = (const float*)d_in[19];
  const float* attn_wv  = (const float*)d_in[20];
  const float* attn_wo  = (const float*)d_in[21];
  const float* attn_mk  = (const float*)d_in[22];
  const float* attn_mv  = (const float*)d_in[23];
  const float* ln_g     = (const float*)d_in[24];
  const float* ln_b     = (const float*)d_in[25];
  const float* out_w    = (const float*)d_in[26];
  const float* out_b    = (const float*)d_in[27];

  float* out = (float*)d_out;

  // workspace layout (floats)
  float* W = (float*)d_ws;
  float* embed   = W;                        // 2048*512
  float* hbuf    = W + 1048576;              // 2048*1024  (also ktmp/vtmp)
  float* cur     = W + 3145728;              // 2048*512
  float* pbuf    = W + 4194304;              // 2048*512   (also oproj)
  float* qbuf    = W + 5242880;              // 2048*512
  float* obuf    = W + 6291456;              // 2048*512
  float* kfull   = W + 7340032;              // 4*576*512
  float* vfull   = W + 8519680;              // 4*576*512
  float* gatev   = W + 9699328;              // 2048
  float* gates4  = W + 9701376;              // 2048*4
  float* auxbuf  = W + 9709568;              // 8 (+pad)
  float* xpad    = W + 9709600;              // 2048*96
  float* ew1pad  = W + 9906208;              // 96*1024
  float* f0w1pad = W + 10004512;             // 4*96*1024
  int* I = (int*)(W + 10397728);
  int* expert_id = I;                        // 2048
  int* idxmap    = I + 2048;                 // 2048
  int* counts    = I + 4096;                 // 4
  int* offsets   = I + 4100;                 // 5
  int* cursor    = I + 4105;                 // 4

  float* ktmp = hbuf;
  float* vtmp = hbuf + 1048576;
  float* oproj = pbuf;

  // ---- pad K=80 operands to K=96 once per call ----
  pad_rows_kernel<<<(KBT * KINP) / 256, 256, 0, stream>>>(x, xpad);
  pad_tail_kernel<<<(KINP * KHID) / 256, 256, 0, stream>>>(emb_w1, ew1pad,
                                                           KIN * KHID, KINP * KHID);
  pad_f0w1_kernel<<<(KE * KINP * KHID) / 256, 256, 0, stream>>>(f0_w1, f0w1pad);

  // ---- embed sub-network + embed_out head ----
  launch_gemm(xpad, ew1pad, emb_b1, hbuf, KBT, KHID, KINP, true, stream);
  launch_gemm(hbuf, emb_w2, nullptr, embed, KBT, KD, KHID, false, stream);
  launch_gemm(embed, emb_wo, nullptr, out + KBTOUT, KBT, KOUT, KD, false, stream);

  int li = 0;
  for (int blk = 0; blk < 2; ++blk) {
    for (int l = 0; l < 4; ++l) {
      const int j = li - 1;
      const float* Rw = (li == 0) ? f0_rout : f_rout + (size_t)j * KD * KE;
      const float* W1 = (li == 0) ? f0w1pad : f_w1 + (size_t)j * KE * KD * KHID;
      const float* B1 = (li == 0) ? f0_b1 : f_b1 + (size_t)j * KE * KHID;
      const float* W2 = (li == 0) ? f0_w2 : f_w2 + (size_t)j * KE * KHID * KD;
      const float* fb = (li == 0) ? f0_fb : f_fb + (size_t)j * KLB * KD;
      const float* fa = (li == 0) ? f0_fa : f_fa + (size_t)j * 1 * KD;
      const float* Xi = (li == 0) ? xpad : cur;
      const int Kin = (li == 0) ? KINP : KD;

      zero_ints_kernel<<<1, 64, 0, stream>>>(counts, 13);
      router_kernel<<<KBT / 8, 256, 0, stream>>>(embed, Rw, expert_id, gatev, gates4,
                                                 counts, seq_len);
      scan_kernel<<<1, 1, 0, stream>>>(counts, offsets);
      scatter_kernel<<<KBT / 256, 256, 0, stream>>>(expert_id, offsets, cursor, idxmap);
      moe_gemm1_kernel<<<dim3(KHID / 64, 8, KE), 256, 0, stream>>>(Xi, W1, B1, idxmap,
                                                                   offsets, hbuf, Kin);
      moe_gemm2_kernel<<<dim3(KD / 64, 8, KE), 256, 0, stream>>>(hbuf, W2, idxmap,
                                                                 offsets, gatev, pbuf);
      fsmn_kernel<<<(KBT * KD) / 256, 256, 0, stream>>>(pbuf, fb, fa, cur, cur,
                                                        (li > 0) ? 1 : 0);
      aux_kernel<<<1, 256, 0, stream>>>(gates4, auxbuf, li);
      ++li;
    }
    if (blk == 0) pe_kernel<<<(KBT * KD) / 256, 256, 0, stream>>>(cur);

    const float* wq = attn_wq + (size_t)blk * KD * KD;
    const float* wk = attn_wk + (size_t)blk * KD * KD;
    const float* wv = attn_wv + (size_t)blk * KD * KD;
    const float* wo = attn_wo + (size_t)blk * KD * KD;
    const float* mk = attn_mk + (size_t)blk * KM * KD;
    const float* mv = attn_mv + (size_t)blk * KM * KD;
    const float* lg = ln_g + (size_t)blk * KD;
    const float* lb = ln_b + (size_t)blk * KD;

    launch_gemm(cur, wq, nullptr, qbuf, KBT, KD, KD, false, stream);
    launch_gemm(cur, wk, nullptr, ktmp, KBT, KD, KD, false, stream);
    launch_gemm(cur, wv, nullptr, vtmp, KBT, KD, KD, false, stream);
    build_kv_kernel<<<(KB * KTM * KD) / 256, 256, 0, stream>>>(ktmp, vtmp, mk, mv,
                                                               kfull, vfull);
    attention_kernel<<<(KB * KH * KT) / 8, 256, 0, stream>>>(qbuf, kfull, vfull,
                                                             seq_len, obuf);
    launch_gemm(obuf, wo, nullptr, oproj, KBT, KD, KD, false, stream);
    residual_ln_kernel<<<KBT / 8, 256, 0, stream>>>(cur, oproj, lg, lb);
  }

  // ---- output head + aux scalar ----
  launch_gemm(cur, out_w, out_b, out, KBT, KOUT, KD, false, stream);
  sum_aux_kernel<<<1, 1, 0, stream>>>(auxbuf, out + 2 * KBTOUT);
}
